// MultiHeadAttention_59416577572874
// MI455X (gfx1250) — matrix operand order repaired and checked by reading, not compiled, not run on hardware
//
#include <hip/hip_runtime.h>
#include <hip/hip_bf16.h>

typedef __attribute__((ext_vector_type(16))) __bf16 v16bf;
typedef __attribute__((ext_vector_type(8)))  __bf16 v8bf;
typedef __attribute__((ext_vector_type(8)))  float  v8f;
typedef __attribute__((ext_vector_type(4)))  float  v4f;

#define D_MODEL 1024
#define NHEAD   16
#define HDIM    64
#define BATCH   4
#define SEQ     2048

static __device__ inline v8f zero8() {
  v8f z;
#pragma unroll
  for (int i = 0; i < 8; ++i) z[i] = 0.0f;
  return z;
}

// A-matrix 16x32 bf16 fragment: lane -> M = lane%16,
// elements 0-15 -> K = 16*(lane/16)+0..15 (contiguous, matches B).
// Caller passes p = &A[m*ldk + kbase + off16]; we read [0..15].
static __device__ inline v16bf frag_a_bf16(const __bf16* p) {
  v8bf lo = *(const v8bf*)(p);
  v8bf hi = *(const v8bf*)(p + 8);
  v16bf a;
#pragma unroll
  for (int i = 0; i < 8; ++i) { a[i] = lo[i]; a[i + 8] = hi[i]; }
  return a;
}

static __device__ inline v16bf frag_a_f32(const float* p) {
  v4f c0 = *(const v4f*)(p);
  v4f c1 = *(const v4f*)(p + 4);
  v4f c2 = *(const v4f*)(p + 8);
  v4f c3 = *(const v4f*)(p + 12);
  v16bf a;
#pragma unroll
  for (int i = 0; i < 4; ++i) {
    a[i]      = (__bf16)c0[i];
    a[i + 4]  = (__bf16)c1[i];
    a[i + 8]  = (__bf16)c2[i];
    a[i + 12] = (__bf16)c3[i];
  }
  return a;
}

static __device__ inline v8f wmma_bf16(v16bf a, v16bf b, v8f c) {
  return __builtin_amdgcn_wmma_f32_16x16x32_bf16(false, a, false, b, (short)0, c,
                                                 false, false);
}

__global__ __launch_bounds__(256) void cvt_f32_bf16(const float* __restrict__ in,
                                                    __bf16* __restrict__ out, int n,
                                                    float scale) {
  int i = blockIdx.x * blockDim.x + threadIdx.x;
  if (i < n) out[i] = (__bf16)(in[i] * scale);
}

// C[m,n] = sum_k A[m,k] * W[n,k] + bias[n]*bias_scale
// MODE 0: A=f32, store bf16 [B,H,S,Dh]          (Q, pre-scaled by 1/sqrt(Dh))
// MODE 1: A=f32, store bf16 [B,H,S,Dh]          (K)
// MODE 2: A=f32, store bf16 [B,H,Dh,S] (V^T)    (V)
// MODE 3: A=bf16 ctx, store f32 [M, D_MODEL]    (out proj)
// Wave tile: 64x64 (16 WMMAs / k-step). Block: 8 waves -> 128x256.
template <int MODE>
__global__ __launch_bounds__(256) void gemm_wmma(
    const float* __restrict__ Af, const __bf16* __restrict__ Ab,
    const __bf16* __restrict__ W, const float* __restrict__ bias, float bias_scale,
    __bf16* __restrict__ outb, float* __restrict__ outf) {
  const int lane  = threadIdx.x & 31;
  const int wid   = threadIdx.x >> 5;
  const int l16   = lane & 15;
  const int hi    = lane >> 4;
  const int off8  = hi * 8;
  const int off16 = hi * 16;

  const int m0 = blockIdx.y * 128 + (wid & 1) * 64;   // 64 rows per wave
  const int n0 = blockIdx.x * 256 + (wid >> 1) * 64;  // 64 cols per wave

  v8f acc[4][4];
#pragma unroll
  for (int i = 0; i < 4; ++i)
#pragma unroll
    for (int j = 0; j < 4; ++j) acc[i][j] = zero8();

  for (int kb = 0; kb < D_MODEL; kb += 32) {
    v16bf a[4];
#pragma unroll
    for (int mt = 0; mt < 4; ++mt) {
      if (MODE == 3)
        a[mt] = frag_a_bf16(Ab + (size_t)(m0 + mt * 16 + l16) * D_MODEL + kb + off16);
      else
        a[mt] = frag_a_f32(Af + (size_t)(m0 + mt * 16 + l16) * D_MODEL + kb + off16);
    }
    v16bf bfrag[4];
#pragma unroll
    for (int nt = 0; nt < 4; ++nt)  // B[k,n] = W[n,k]: lane n = l16, elems k contig
      bfrag[nt] = *(const v16bf*)(W + (size_t)(n0 + nt * 16 + l16) * D_MODEL + kb + off16);
#pragma unroll
    for (int mt = 0; mt < 4; ++mt)
#pragma unroll
      for (int nt = 0; nt < 4; ++nt)
        acc[mt][nt] = wmma_bf16(a[mt], bfrag[nt], acc[mt][nt]);
  }

  // C/D layout: VGPR r -> M = r + 8*(lane/16); N = lane%16
#pragma unroll
  for (int mt = 0; mt < 4; ++mt) {
#pragma unroll
    for (int nt = 0; nt < 4; ++nt) {
      const int n = n0 + nt * 16 + l16;
      const float bn = bias[n] * bias_scale;
#pragma unroll
      for (int r = 0; r < 8; ++r) {
        const int m = m0 + mt * 16 + r + 8 * hi;
        const float v = acc[mt][nt][r] + bn;
        if (MODE == 3) {
          outf[(size_t)m * D_MODEL + n] = v;
        } else {
          const int b = m >> 11, s = m & (SEQ - 1);
          const int h = n >> 6, d = n & (HDIM - 1);
          size_t idx;
          if (MODE == 2) idx = (((size_t)(b * NHEAD + h)) * HDIM + d) * SEQ + s;
          else           idx = (((size_t)(b * NHEAD + h)) * SEQ + s) * HDIM + d;
          outb[idx] = (__bf16)v;
        }
      }
    }
  }
}

// Causal flash attention. Q (pre-scaled), K: [B,H,S,Dh] bf16; Vt: [B,H,Dh,S] bf16;
// Ctx: [B,S,D] bf16. Block = 4 waves; wave = 16 query rows; 64 keys per iteration.
__global__ __launch_bounds__(128) void attn_wmma(const __bf16* __restrict__ Q,
                                                 const __bf16* __restrict__ K,
                                                 const __bf16* __restrict__ Vt,
                                                 __bf16* __restrict__ Ctx) {
  __shared__ __align__(64) __bf16 plds[4][16 * 64];
  const int lane  = threadIdx.x & 31;
  const int wid   = threadIdx.x >> 5;
  const int l16   = lane & 15;
  const int hi    = lane >> 4;
  const int off8  = hi * 8;
  const int off16 = hi * 16;

  const int bh = blockIdx.y;
  const int b = bh >> 4, h = bh & 15;
  const __bf16* Qh = Q + (size_t)bh * SEQ * HDIM;
  const __bf16* Kh = K + (size_t)bh * SEQ * HDIM;
  const __bf16* Vh = Vt + (size_t)bh * HDIM * SEQ;

  const int q0 = blockIdx.x * 64 + wid * 16;  // this wave's query rows

  // Q A-fragments: dh 0..31 and 32..63 (already scaled by 1/sqrt(Dh))
  const v16bf qa0 = frag_a_bf16(Qh + (size_t)(q0 + l16) * HDIM + 0 + off16);
  const v16bf qa1 = frag_a_bf16(Qh + (size_t)(q0 + l16) * HDIM + 32 + off16);

  v8f acc[4];
#pragma unroll
  for (int t = 0; t < 4; ++t) acc[t] = zero8();
  float mrow[8], lrow[8];
#pragma unroll
  for (int r = 0; r < 8; ++r) { mrow[r] = -3.0e38f; lrow[r] = 0.0f; }

  for (int kb = 0; kb < q0 + 16; kb += 64) {
    float p[4][8];
    const bool need_mask = (kb + 63 > q0);
#pragma unroll
    for (int g = 0; g < 4; ++g) {
      // S = Q * K^T: B[dh, key] = K[key, dh] -> lane = key, elems = dh (contig)
      v16bf blo = *(const v16bf*)(Kh + (size_t)(kb + g * 16 + l16) * HDIM + 0 + off16);
      v16bf bhi = *(const v16bf*)(Kh + (size_t)(kb + g * 16 + l16) * HDIM + 32 + off16);
      v8f s = wmma_bf16(qa0, blo, zero8());
      s = wmma_bf16(qa1, bhi, s);
      const int key = kb + g * 16 + l16;
#pragma unroll
      for (int r = 0; r < 8; ++r) {
        float sv = s[r];
        if (need_mask) {
          const int qrow = q0 + r + 8 * hi;
          if (key > qrow) sv = -1.0e30f;
        }
        p[g][r] = sv;
      }
    }
    if (kb + 64 < SEQ)  // speculative prefetch of next K block (global_prefetch_b8)
      __builtin_prefetch(Kh + (size_t)(kb + 64 + lane) * HDIM, 0, 1);
    // Online softmax: row m = r + 8*hi lives across the 16-lane half-group.
#pragma unroll
    for (int r = 0; r < 8; ++r) {
      float rm = fmaxf(fmaxf(p[0][r], p[1][r]), fmaxf(p[2][r], p[3][r]));
      rm = fmaxf(rm, __shfl_xor(rm, 1));
      rm = fmaxf(rm, __shfl_xor(rm, 2));
      rm = fmaxf(rm, __shfl_xor(rm, 4));
      rm = fmaxf(rm, __shfl_xor(rm, 8));
      const float mnew = fmaxf(mrow[r], rm);
      const float alpha = __expf(mrow[r] - mnew);
      mrow[r] = mnew;
      float rs = 0.0f;
#pragma unroll
      for (int g = 0; g < 4; ++g) {
        const float pe = __expf(p[g][r] - mnew);
        p[g][r] = pe;
        rs += pe;
      }
      rs += __shfl_xor(rs, 1);
      rs += __shfl_xor(rs, 2);
      rs += __shfl_xor(rs, 4);
      rs += __shfl_xor(rs, 8);
      lrow[r] = lrow[r] * alpha + rs;
#pragma unroll
      for (int t = 0; t < 4; ++t) acc[t][r] = acc[t][r] * alpha;
    }
    // Stage P (D-layout) through LDS to read back in A-layout (16x64, row stride 64)
    __bf16* pw = &plds[wid][0];
#pragma unroll
    for (int g = 0; g < 4; ++g)
#pragma unroll
      for (int r = 0; r < 8; ++r)
        pw[(r + 8 * hi) * 64 + g * 16 + l16] = (__bf16)p[g][r];
    asm volatile("s_wait_dscnt 0" ::: "memory");
    const v16bf pa0 = frag_a_bf16(pw + l16 * 64 + 0 + off16);
    const v16bf pa1 = frag_a_bf16(pw + l16 * 64 + 32 + off16);
    // acc += P * V: B[key, dh] = Vt[dh, key] -> lane = dh, elems = key (contig)
#pragma unroll
    for (int t = 0; t < 4; ++t) {
      v16bf vb0 = *(const v16bf*)(Vh + (size_t)(t * 16 + l16) * SEQ + kb + 0 + off16);
      v16bf vb1 = *(const v16bf*)(Vh + (size_t)(t * 16 + l16) * SEQ + kb + 32 + off16);
      acc[t] = wmma_bf16(pa0, vb0, acc[t]);
      acc[t] = wmma_bf16(pa1, vb1, acc[t]);
    }
  }

#pragma unroll
  for (int r = 0; r < 8; ++r) {
    const float inv = 1.0f / lrow[r];
    const int s = q0 + r + 8 * hi;
#pragma unroll
    for (int t = 0; t < 4; ++t)
      Ctx[((size_t)b * SEQ + s) * D_MODEL + h * HDIM + t * 16 + l16] =
          (__bf16)(acc[t][r] * inv);
  }
}

extern "C" void kernel_launch(void* const* d_in, const int* in_sizes, int n_in,
                              void* d_out, int out_size, void* d_ws, size_t ws_size,
                              hipStream_t stream) {
  const float* q  = (const float*)d_in[0];
  const float* k  = (const float*)d_in[1];
  const float* v  = (const float*)d_in[2];
  const float* Wq = (const float*)d_in[4];
  const float* bq = (const float*)d_in[5];
  const float* Wk = (const float*)d_in[6];
  const float* bk = (const float*)d_in[7];
  const float* Wv = (const float*)d_in[8];
  const float* bv = (const float*)d_in[9];
  const float* Wo = (const float*)d_in[10];
  const float* bo = (const float*)d_in[11];
  (void)in_sizes; (void)n_in; (void)out_size; (void)ws_size;

  char* ws = (char*)d_ws;
  const size_t MB = (size_t)1 << 20;
  __bf16* wqb = (__bf16*)(ws + 0 * MB);
  __bf16* wkb = (__bf16*)(ws + 2 * MB);
  __bf16* wvb = (__bf16*)(ws + 4 * MB);
  __bf16* wob = (__bf16*)(ws + 6 * MB);
  __bf16* Qb  = (__bf16*)(ws + 8 * MB);   // [B,H,S,Dh], pre-scaled
  __bf16* Kb  = (__bf16*)(ws + 24 * MB);  // [B,H,S,Dh]
  __bf16* Vtb = (__bf16*)(ws + 40 * MB);  // [B,H,Dh,S]
  __bf16* Ctx = (__bf16*)(ws + 56 * MB);  // [B*S, D]

  const float qscale = 0.125f;  // 1/sqrt(HDIM), folded into Wq/bq
  const int NW = D_MODEL * D_MODEL;
  cvt_f32_bf16<<<NW / 256, 256, 0, stream>>>(Wq, wqb, NW, qscale);
  cvt_f32_bf16<<<NW / 256, 256, 0, stream>>>(Wk, wkb, NW, 1.0f);
  cvt_f32_bf16<<<NW / 256, 256, 0, stream>>>(Wv, wvb, NW, 1.0f);
  cvt_f32_bf16<<<NW / 256, 256, 0, stream>>>(Wo, wob, NW, 1.0f);

  dim3 gg(D_MODEL / 256, (BATCH * SEQ) / 128);
  gemm_wmma<0><<<gg, 256, 0, stream>>>(q, nullptr, wqb, bq, qscale, Qb, nullptr);
  gemm_wmma<1><<<gg, 256, 0, stream>>>(k, nullptr, wkb, bk, 1.0f, Kb, nullptr);
  gemm_wmma<2><<<gg, 256, 0, stream>>>(v, nullptr, wvb, bv, 1.0f, Vtb, nullptr);

  attn_wmma<<<dim3(SEQ / 64, BATCH * NHEAD), 128, 0, stream>>>(Qb, Kb, Vtb, Ctx);

  gemm_wmma<3><<<gg, 256, 0, stream>>>(nullptr, Ctx, wob, bo, 1.0f, nullptr,
                                       (float*)d_out);
}
